// MHA_71975061946783
// MI455X (gfx1250) — matrix operand order repaired and checked by reading, not compiled, not run on hardware
//
#include <hip/hip_runtime.h>
#include <hip/hip_bf16.h>

typedef __bf16 bf16;
typedef __attribute__((ext_vector_type(16))) __bf16 v16bf;
typedef __attribute__((ext_vector_type(8)))  __bf16 v8bf;
typedef __attribute__((ext_vector_type(4)))  __bf16 v4bf;
typedef __attribute__((ext_vector_type(8)))  float  v8f;

union Frag16 { v16bf v; v8bf h[2]; };

__device__ __forceinline__ v8f wmma_bf16(const Frag16& a, const Frag16& b, v8f c) {
    // D = A(16x32 bf16) x B(32x16 bf16) + C(16x16 f32)
    return __builtin_amdgcn_wmma_f32_16x16x32_bf16(
        /*neg_a=*/false, a.v, /*neg_b=*/false, b.v,
        /*c_mod=*/(short)0, c, /*reuse_a=*/false, /*reuse_b=*/false);
}

// ---------------------------------------------------------------------------
// fp32 -> bf16, 4 elements / thread
// ---------------------------------------------------------------------------
__global__ void cvt_bf16(const float* __restrict__ in, bf16* __restrict__ out, int n4) {
    int i = blockIdx.x * blockDim.x + threadIdx.x;
    if (i < n4) {
        float4 f = ((const float4*)in)[i];
        v4bf o = { (bf16)f.x, (bf16)f.y, (bf16)f.z, (bf16)f.w };
        ((v4bf*)out)[i] = o;
    }
}

// ---------------------------------------------------------------------------
// fp32 W[K][N] -> bf16 Wt[N][K]  (one-time transposed weight conversion)
// grid (N/32, K/32), block 256; 32x33 LDS tile avoids bank conflicts.
// ---------------------------------------------------------------------------
__global__ void cvt_transpose_bf16(const float* __restrict__ W, bf16* __restrict__ Wt,
                                   int K, int N) {
    __shared__ bf16 tile[32][33];
    const int k0 = blockIdx.y * 32, n0 = blockIdx.x * 32;
    const int c  = threadIdx.x & 31;
    const int r0 = threadIdx.x >> 5;      // 0..7
#pragma unroll
    for (int i = 0; i < 4; ++i) {
        int r = r0 + i * 8;
        tile[r][c] = (bf16)W[(size_t)(k0 + r) * N + n0 + c];
    }
    __syncthreads();
#pragma unroll
    for (int i = 0; i < 4; ++i) {
        int r = r0 + i * 8;
        Wt[(size_t)(n0 + r) * K + k0 + c] = tile[c][r];
    }
}

// ---------------------------------------------------------------------------
// C = A(MxK bf16, row-major) @ B + bias, with B given pre-transposed as
// BT[N][K] bf16. Block tile 128x128, 8 waves, wave tile 32x64
// (2x4 WMMA 16x16x32 bf16 accumulators = 8 WMMA / K-step / wave).
// Double-buffered LDS: one barrier per K-step; next-step global loads are
// issued before the WMMAs and stored to the alternate buffer after them.
// MODE: 0 = bf16 row-major out, 1 = bf16 transposed out (Cout[N][M]),
//       2 = f32 row-major out.
// ---------------------------------------------------------------------------
template <int MODE>
__global__ void gemm_bf16_wmma(const bf16* __restrict__ A, const bf16* __restrict__ BT,
                               const float* __restrict__ bias, void* __restrict__ Cout,
                               int M, int N, int K) {
    __shared__ __align__(16) bf16 As[2][128 * 32];   // [buf][row][k]
    __shared__ __align__(16) bf16 Bs[2][128 * 32];   // [buf][n][k]

    const int tid  = threadIdx.x;
    const int lane = tid & 31;
    const int wid  = tid >> 5;          // 0..7
    const int wm   = wid >> 1;          // 0..3 -> M offset wm*32
    const int wn   = wid & 1;           // 0..1 -> N offset wn*64
    const int m0   = blockIdx.y * 128;
    const int n0   = blockIdx.x * 128;
    const int l16  = lane & 15;
    const bool lo  = lane < 16;

    const int srow = tid >> 2;          // 0..63
    const int sseg = (tid & 3) * 8;     // 0,8,16,24

    v8bf a0, a1, b0, b1;
    // prologue: stage K-step 0 into buffer 0
    a0 = *(const v8bf*)&A [(size_t)(m0 + srow)      * K + sseg];
    a1 = *(const v8bf*)&A [(size_t)(m0 + srow + 64) * K + sseg];
    b0 = *(const v8bf*)&BT[(size_t)(n0 + srow)      * K + sseg];
    b1 = *(const v8bf*)&BT[(size_t)(n0 + srow + 64) * K + sseg];
    *(v8bf*)&As[0][srow * 32 + sseg]        = a0;
    *(v8bf*)&As[0][(srow + 64) * 32 + sseg] = a1;
    *(v8bf*)&Bs[0][srow * 32 + sseg]        = b0;
    *(v8bf*)&Bs[0][(srow + 64) * 32 + sseg] = b1;
    __syncthreads();

    v8f acc[2][4] = {};

    for (int kk = 0, it = 0; kk < K; kk += 32, ++it) {
        const int  cur  = it & 1;
        const int  nxt  = cur ^ 1;
        const bool more = (kk + 32 < K);

        if (more) {   // issue next-step global loads; latency hidden by WMMAs
            a0 = *(const v8bf*)&A [(size_t)(m0 + srow)      * K + kk + 32 + sseg];
            a1 = *(const v8bf*)&A [(size_t)(m0 + srow + 64) * K + kk + 32 + sseg];
            b0 = *(const v8bf*)&BT[(size_t)(n0 + srow)      * K + kk + 32 + sseg];
            b1 = *(const v8bf*)&BT[(size_t)(n0 + srow + 64) * K + kk + 32 + sseg];
        }

        Frag16 af[2], bfm[4];
#pragma unroll
        for (int f = 0; f < 2; ++f) {
            int r  = wm * 32 + f * 16 + l16;
            int ab = r * 32 + (lo ? 0 : 8);       // A layout: K 0-7/8-15, 16-23/24-31
            af[f].h[0] = *(const v8bf*)&As[cur][ab];
            af[f].h[1] = *(const v8bf*)&As[cur][ab + 16];
        }
#pragma unroll
        for (int f = 0; f < 4; ++f) {
            int n  = wn * 64 + f * 16 + l16;
            int bb = n * 32 + (lo ? 0 : 8);       // B layout: K 0-7/8-15, 16-23/24-31
            bfm[f].h[0] = *(const v8bf*)&Bs[cur][bb];
            bfm[f].h[1] = *(const v8bf*)&Bs[cur][bb + 16];
        }
#pragma unroll
        for (int i = 0; i < 2; ++i)
#pragma unroll
            for (int j = 0; j < 4; ++j)
                acc[i][j] = wmma_bf16(af[i], bfm[j], acc[i][j]);

        if (more) {   // stash prefetched tile into the alternate buffer
            *(v8bf*)&As[nxt][srow * 32 + sseg]        = a0;
            *(v8bf*)&As[nxt][(srow + 64) * 32 + sseg] = a1;
            *(v8bf*)&Bs[nxt][srow * 32 + sseg]        = b0;
            *(v8bf*)&Bs[nxt][(srow + 64) * 32 + sseg] = b1;
        }
        __syncthreads();
    }

#pragma unroll
    for (int i = 0; i < 2; ++i) {
#pragma unroll
        for (int j = 0; j < 4; ++j) {
            int   n  = n0 + wn * 64 + j * 16 + l16;
            float bv = bias[n];
            if constexpr (MODE == 1) {
                // transposed bf16 out: 8 consecutive M per lane -> one b128 store
                v8bf pk;
#pragma unroll
                for (int r = 0; r < 8; ++r) pk[r] = (bf16)(acc[i][j][r] + bv);
                int mb = m0 + wm * 32 + i * 16 + (lo ? 0 : 8);
                *(v8bf*)&((bf16*)Cout)[(size_t)n * M + mb] = pk;
            } else {
#pragma unroll
                for (int r = 0; r < 8; ++r) {
                    int   m   = m0 + wm * 32 + i * 16 + (lo ? r : r + 8);
                    float val = acc[i][j][r] + bv;
                    if constexpr (MODE == 2)
                        ((float*)Cout)[(size_t)m * N + n] = val;
                    else
                        ((bf16*)Cout)[(size_t)m * N + n] = (bf16)val;
                }
            }
        }
    }
}

// ---------------------------------------------------------------------------
// Causal flash attention. Grid (T/64, B*H), block 128 (4 waves).
// Each wave: 16 q-rows, online softmax, WMMA for QK^T and PV.
// Q/K layout: [B*T][C] bf16 (head h = cols h*64..h*64+63).
// V layout:   pre-transposed VT[C][B*T] bf16 (row = h*64+d, col = token).
// Double-buffered K/V tiles: one barrier per key-tile.
// ---------------------------------------------------------------------------
__global__ void attn_wmma(const bf16* __restrict__ Q, const bf16* __restrict__ Kb,
                          const bf16* __restrict__ VT, bf16* __restrict__ Y) {
    constexpr int T = 2048, C = 1024, MTOT = 8192;
    __shared__ __align__(16) bf16 Ks[2][32 * 64];   // [buf][key][hs]
    __shared__ __align__(16) bf16 Vts[2][64 * 32];  // [buf][hs][key]
    __shared__ __align__(16) bf16 Ps[4][16 * 32];   // per-wave P staging [m][key]

    const int tid  = threadIdx.x;
    const int lane = tid & 31;
    const int wid  = tid >> 5;      // 0..3
    const int l16  = lane & 15;
    const bool lo  = lane < 16;
    const int qt   = blockIdx.x;    // q tile of 64 rows
    const int bh   = blockIdx.y;
    const int b    = bh >> 4, h = bh & 15;
    const size_t tokbase = (size_t)b * T;
    const int qrow0 = qt * 64 + wid * 16;

    // Q fragments: 2 hs-chunks of 32
    Frag16 qf[2];
    {
        size_t qr = (tokbase + qrow0 + l16) * C + h * 64;
#pragma unroll
        for (int c = 0; c < 2; ++c) {
            size_t base = qr + c * 32 + (lo ? 0 : 8);
            qf[c].h[0] = *(const v8bf*)&Q[base];
            qf[c].h[1] = *(const v8bf*)&Q[base + 16];
        }
    }

    v8f accO[4] = {};
    float mrun[8], lrun[8];
#pragma unroll
    for (int r = 0; r < 8; ++r) { mrun[r] = -3.0e38f; lrun[r] = 0.0f; }

    const int skey = tid >> 2;          // 0..31  (K staging)
    const int sseg = (tid & 3) * 8;     // 0,8,16,24
    const int vrow = tid >> 1;          // 0..63  (V staging, already transposed)
    const int vseg = (tid & 1) * 16;    // 0,16
    const float sm_scale = 0.125f;      // 1/sqrt(64)
    const int ntiles = qt * 2 + 2;      // keys up to (qt+1)*64

    v8bf kr0, kr1, vr0, vr1;
    {   // prologue: stage key-tile 0 into buffer 0
        size_t gk = (tokbase + skey) * C + h * 64;
        kr0 = *(const v8bf*)&Kb[gk + sseg];
        kr1 = *(const v8bf*)&Kb[gk + sseg + 32];
        size_t gv = (size_t)(h * 64 + vrow) * MTOT + tokbase + vseg;
        vr0 = *(const v8bf*)&VT[gv];
        vr1 = *(const v8bf*)&VT[gv + 8];
        *(v8bf*)&Ks[0][skey * 64 + sseg]       = kr0;
        *(v8bf*)&Ks[0][skey * 64 + sseg + 32]  = kr1;
        *(v8bf*)&Vts[0][vrow * 32 + vseg]      = vr0;
        *(v8bf*)&Vts[0][vrow * 32 + vseg + 8]  = vr1;
    }
    __syncthreads();

    for (int kt = 0; kt < ntiles; ++kt) {
        const int  cur  = kt & 1;
        const int  nxt  = cur ^ 1;
        const bool more = (kt + 1 < ntiles);
        const int  key0 = kt * 32;

        if (more) {   // issue next-tile global loads; hidden by compute below
            size_t gk = (tokbase + key0 + 32 + skey) * C + h * 64;
            kr0 = *(const v8bf*)&Kb[gk + sseg];
            kr1 = *(const v8bf*)&Kb[gk + sseg + 32];
            size_t gv = (size_t)(h * 64 + vrow) * MTOT + tokbase + key0 + 32 + vseg;
            vr0 = *(const v8bf*)&VT[gv];
            vr1 = *(const v8bf*)&VT[gv + 8];
        }

        // S = (Q K^T) * scale, two 16-key subtiles, causal mask
        float p[2][8];
#pragma unroll
        for (int st = 0; st < 2; ++st) {
            Frag16 kf[2];
            int key = st * 16 + l16;
#pragma unroll
            for (int c = 0; c < 2; ++c) {
                int base = key * 64 + c * 32 + (lo ? 0 : 8);
                kf[c].h[0] = *(const v8bf*)&Ks[cur][base];
                kf[c].h[1] = *(const v8bf*)&Ks[cur][base + 16];
            }
            v8f s = {};
            s = wmma_bf16(qf[0], kf[0], s);
            s = wmma_bf16(qf[1], kf[1], s);
            int kj = key0 + st * 16 + l16;
#pragma unroll
            for (int r = 0; r < 8; ++r) {
                int qi = qrow0 + (lo ? r : r + 8);
                p[st][r] = (kj <= qi) ? s[r] * sm_scale : -3.0e38f;
            }
        }

        // online softmax: row stats via shfl within 16-lane halves
#pragma unroll
        for (int r = 0; r < 8; ++r) {
            float mx = fmaxf(p[0][r], p[1][r]);
#pragma unroll
            for (int off = 8; off >= 1; off >>= 1)
                mx = fmaxf(mx, __shfl_xor(mx, off, 32));
            float mnew = fmaxf(mrun[r], mx);
            float scl  = __expf(mrun[r] - mnew);
            float p0   = __expf(p[0][r] - mnew);
            float p1   = __expf(p[1][r] - mnew);
            p[0][r] = p0; p[1][r] = p1;
            float sum = p0 + p1;
#pragma unroll
            for (int off = 8; off >= 1; off >>= 1)
                sum += __shfl_xor(sum, off, 32);
            lrun[r] = lrun[r] * scl + sum;
            mrun[r] = mnew;
#pragma unroll
            for (int c = 0; c < 4; ++c) accO[c][r] *= scl;
        }

        // C-layout -> A-layout via per-wave LDS round trip
#pragma unroll
        for (int st = 0; st < 2; ++st)
#pragma unroll
            for (int r = 0; r < 8; ++r) {
                int m = lo ? r : r + 8;
                Ps[wid][m * 32 + st * 16 + l16] = (bf16)p[st][r];
            }
        Frag16 pa;
        {
            int base = l16 * 32 + (lo ? 0 : 8);
            pa.h[0] = *(const v8bf*)&Ps[wid][base];
            pa.h[1] = *(const v8bf*)&Ps[wid][base + 16];
        }
        // O += P @ V   (4 hs-column chunks of 16)
#pragma unroll
        for (int c = 0; c < 4; ++c) {
            Frag16 vf;
            int base = (c * 16 + l16) * 32 + (lo ? 0 : 8);
            vf.h[0] = *(const v8bf*)&Vts[cur][base];
            vf.h[1] = *(const v8bf*)&Vts[cur][base + 16];
            accO[c] = wmma_bf16(pa, vf, accO[c]);
        }

        if (more) {   // stash prefetched tile into the alternate buffer
            *(v8bf*)&Ks[nxt][skey * 64 + sseg]       = kr0;
            *(v8bf*)&Ks[nxt][skey * 64 + sseg + 32]  = kr1;
            *(v8bf*)&Vts[nxt][vrow * 32 + vseg]      = vr0;
            *(v8bf*)&Vts[nxt][vrow * 32 + vseg + 8]  = vr1;
        }
        __syncthreads();
    }

    // y = O / l
#pragma unroll
    for (int r = 0; r < 8; ++r) {
        float inv = 1.0f / lrun[r];
        int m = lo ? r : r + 8;
        size_t row = (tokbase + qrow0 + m) * C + h * 64;
#pragma unroll
        for (int c = 0; c < 4; ++c)
            Y[row + c * 16 + l16] = (bf16)(accO[c][r] * inv);
    }
}

// ---------------------------------------------------------------------------
extern "C" void kernel_launch(void* const* d_in, const int* in_sizes, int n_in,
                              void* d_out, int out_size, void* d_ws, size_t ws_size,
                              hipStream_t stream) {
    const float* x  = (const float*)d_in[0];
    const float* Wq = (const float*)d_in[1];
    const float* bq = (const float*)d_in[2];
    const float* Wk = (const float*)d_in[3];
    const float* bk = (const float*)d_in[4];
    const float* Wv = (const float*)d_in[5];
    const float* bv = (const float*)d_in[6];
    const float* Wo = (const float*)d_in[7];
    const float* bo = (const float*)d_in[8];

    const int Btok = 4 * 2048;   // 8192 token rows
    const int Cdim = 1024;

    size_t off = 0;
    auto carve = [&](size_t bytes) {
        void* p = (char*)d_ws + off;
        off += (bytes + 255) & ~(size_t)255;
        return p;
    };
    bf16* xbf  = (bf16*)carve((size_t)Btok * Cdim * 2);
    bf16* wqt  = (bf16*)carve((size_t)Cdim * Cdim * 2);   // [N][K]
    bf16* wkt  = (bf16*)carve((size_t)Cdim * Cdim * 2);
    bf16* wvt  = (bf16*)carve((size_t)Cdim * Cdim * 2);
    bf16* wot  = (bf16*)carve((size_t)Cdim * Cdim * 2);
    bf16* qb   = (bf16*)carve((size_t)Btok * Cdim * 2);   // [tok][C]
    bf16* kb   = (bf16*)carve((size_t)Btok * Cdim * 2);   // [tok][C]
    bf16* vtb  = (bf16*)carve((size_t)Btok * Cdim * 2);   // [C][tok] (transposed)
    bf16* yb   = (bf16*)carve((size_t)Btok * Cdim * 2);   // [tok][C]

    {   // fp32 -> bf16 (x) and fp32 -> bf16 transposed (weights)
        int n4 = Btok * Cdim / 4;
        cvt_bf16<<<(n4 + 255) / 256, 256, 0, stream>>>(x, xbf, n4);
        dim3 tgrid(Cdim / 32, Cdim / 32);
        cvt_transpose_bf16<<<tgrid, 256, 0, stream>>>(Wq, wqt, Cdim, Cdim);
        cvt_transpose_bf16<<<tgrid, 256, 0, stream>>>(Wk, wkt, Cdim, Cdim);
        cvt_transpose_bf16<<<tgrid, 256, 0, stream>>>(Wv, wvt, Cdim, Cdim);
        cvt_transpose_bf16<<<tgrid, 256, 0, stream>>>(Wo, wot, Cdim, Cdim);
    }

    dim3 ggrid(Cdim / 128, Btok / 128);   // (8, 64)
    gemm_bf16_wmma<0><<<ggrid, 256, 0, stream>>>(xbf, wqt, bq, qb,  Btok, Cdim, Cdim);
    gemm_bf16_wmma<0><<<ggrid, 256, 0, stream>>>(xbf, wkt, bk, kb,  Btok, Cdim, Cdim);
    gemm_bf16_wmma<1><<<ggrid, 256, 0, stream>>>(xbf, wvt, bv, vtb, Btok, Cdim, Cdim);

    dim3 agrid(2048 / 64, 4 * 16);        // (32, 64)
    attn_wmma<<<agrid, 128, 0, stream>>>(qb, kb, vtb, yb);

    gemm_bf16_wmma<2><<<ggrid, 256, 0, stream>>>(yb, wot, bo, d_out, Btok, Cdim, Cdim);
}
